// ST_LSTM_37881611550996
// MI455X (gfx1250) — matrix operand order repaired and checked by reading, not compiled, not run on hardware
//
#include <hip/hip_runtime.h>

// ---------------------------------------------------------------------------
// ST-LSTM for MI455X (gfx1250, wave32, WMMA).
// GEMMs run on v_wmma_f32_16x16x32_bf16 (fp32 accumulate). Intermediates are
// stored bf16 so the full working set (~160MB) fits in the 192MB L2.
// LN statistics accumulated in fp32 from WMMA accumulators (exact).
// ---------------------------------------------------------------------------

#define N_BATCH 64
#define C_DIM   128
#define V_LEN   512
#define CV      65536                 // C_DIM * V_LEN
#define NCV_SZ  (4194304u)            // N_BATCH * CV

typedef __attribute__((ext_vector_type(16))) __bf16 v16bf;
typedef __attribute__((ext_vector_type(8)))  float  v8f;

union FragU { v16bf v; uint4 q[2]; };

__device__ __forceinline__ unsigned short f2bf(float f) {
  unsigned u = __float_as_uint(f);
  unsigned r = u + 0x7FFFu + ((u >> 16) & 1u);   // round-to-nearest-even
  return (unsigned short)(r >> 16);
}
__device__ __forceinline__ unsigned pack2bf(float lo, float hi) {
  return ((unsigned)f2bf(lo)) | (((unsigned)f2bf(hi)) << 16);
}
__device__ __forceinline__ float bf2f(unsigned short s) {
  return __uint_as_float(((unsigned)s) << 16);
}
__device__ __forceinline__ float sigm(float x) { return 1.0f / (1.0f + expf(-x)); }

#define LSTR 136    // LDS row stride (bf16 elems): 272B rows, 16B aligned, 4-bank shift/row

// ---------------------------------------------------------------------------
// GEMM + bias + LN-stat epilogue. Block computes a 128(Cout) x 64(V) tile of
// one sample for up to 7 projections sharing the same input (B reused in LDS).
// Full 128x128 weight tile staged once per projection -> 2 barriers/proj.
// ---------------------------------------------------------------------------
struct GemmLnArgs {
  const void* inp; int in_bf16; int nproj;
  const float* w[7]; const float* bias[7];
  unsigned short* z[7];
  float* sums;           // [17][2][64]  (sum, sumsq) per (proj, n)
  int pid[7];
};

__global__ __launch_bounds__(256) void stlstm_gemm_ln(GemmLnArgs a) {
  __shared__ unsigned short Bs[64 * LSTR];    // [v][k]   17408 B
  __shared__ unsigned short As[128 * LSTR];   // [m][k]   34816 B
  const int tid = threadIdx.x;
  const int n   = blockIdx.y;
  const int v0  = blockIdx.x * 64;

  // Stage B tile (K=128 x V=64) transposed into Bs[v][k]; two k packed per b32.
  if (a.in_bf16) {
    const unsigned short* ip = (const unsigned short*)a.inp + (size_t)n * CV + v0;
    #pragma unroll
    for (int it = 0; it < 16; ++it) {
      int i = tid + it * 256;                 // 0..4095
      int v = i & 63, kp = i >> 6;            // kp = k/2
      unsigned short s0 = ip[(2 * kp) * V_LEN + v];
      unsigned short s1 = ip[(2 * kp + 1) * V_LEN + v];
      *(unsigned*)&Bs[v * LSTR + 2 * kp] = (unsigned)s0 | ((unsigned)s1 << 16);
    }
  } else {
    const float* ip = (const float*)a.inp + (size_t)n * CV + v0;
    #pragma unroll
    for (int it = 0; it < 16; ++it) {
      int i = tid + it * 256;
      int v = i & 63, kp = i >> 6;
      float f0 = ip[(2 * kp) * V_LEN + v];
      float f1 = ip[(2 * kp + 1) * V_LEN + v];
      *(unsigned*)&Bs[v * LSTR + 2 * kp] = pack2bf(f0, f1);
    }
  }

  const int wv = tid >> 5, lane = tid & 31;
  const int wm = (wv >> 1) * 32;     // 8 waves: 4 (M) x 2 (N)
  const int wn = (wv & 1) * 32;
  const int l16 = lane & 15, lhi = lane >> 4;
  const v8f zero8 = {0.f, 0.f, 0.f, 0.f, 0.f, 0.f, 0.f, 0.f};

  for (int p = 0; p < a.nproj; ++p) {
    __syncthreads();                          // B ready / prev-proj A reads done
    if (p + 1 < a.nproj)
      __builtin_prefetch(a.w[p + 1] + (tid << 6), 0, 1);

    // Stage full A tile (128 x 128) of weights, packed bf16 pairs.
    const float2* W2 = (const float2*)a.w[p];
    #pragma unroll
    for (int it = 0; it < 32; ++it) {
      int i = tid + it * 256;                 // 0..8191
      int m = i >> 6, kp = i & 63;
      float2 f = W2[m * 64 + kp];
      *(unsigned*)&As[m * LSTR + 2 * kp] = pack2bf(f.x, f.y);
    }
    __syncthreads();

    v8f acc[2][2];
    #pragma unroll
    for (int tm = 0; tm < 2; ++tm)
      #pragma unroll
      for (int tn = 0; tn < 2; ++tn) acc[tm][tn] = zero8;

    #pragma unroll
    for (int ks = 0; ks < 4; ++ks) {
      FragU af[2], bf[2];
      #pragma unroll
      for (int t = 0; t < 2; ++t) {
        // A frag: lanes 0-15 row m, K {0..7,16..23}; lanes 16-31 K {8..15,24..31}
        const unsigned short* ap = &As[(wm + t * 16 + l16) * LSTR + ks * 32 + lhi * 8];
        af[t].q[0] = *(const uint4*)(ap);
        af[t].q[1] = *(const uint4*)(ap + 16);
        // B frag: lane column n, lanes 0-15 K {0..7,16..23}; lanes 16-31 K {8..15,24..31}
        const unsigned short* bp = &Bs[(wn + t * 16 + l16) * LSTR + ks * 32 + lhi * 8];
        bf[t].q[0] = *(const uint4*)(bp);
        bf[t].q[1] = *(const uint4*)(bp + 16);
      }
      #pragma unroll
      for (int tm = 0; tm < 2; ++tm)
        #pragma unroll
        for (int tn = 0; tn < 2; ++tn)
          acc[tm][tn] = __builtin_amdgcn_wmma_f32_16x16x32_bf16(
              false, af[tm].v, false, bf[tn].v, (short)0, acc[tm][tn], false, false);
    }

    // Epilogue: bias, bf16 store, LN partial sums.
    const float* bp_ = a.bias[p];
    unsigned short* zo = a.z[p];
    float lsum = 0.f, lsq = 0.f;
    #pragma unroll
    for (int tm = 0; tm < 2; ++tm) {
      #pragma unroll
      for (int tn = 0; tn < 2; ++tn) {
        int col = v0 + wn + tn * 16 + l16;
        #pragma unroll
        for (int r = 0; r < 8; ++r) {
          int m = wm + tm * 16 + r + lhi * 8;   // C/D layout: lanes>=16 hold M+8
          float val = acc[tm][tn][r] + bp_[m];
          zo[(size_t)n * CV + m * V_LEN + col] = f2bf(val);
          lsum += val; lsq += val * val;
        }
      }
    }
    for (int off = 16; off >= 1; off >>= 1) {
      lsum += __shfl_xor(lsum, off, 32);
      lsq  += __shfl_xor(lsq,  off, 32);
    }
    if (lane == 0) {
      atomicAdd(a.sums + a.pid[p] * 128 + n, lsum);
      atomicAdd(a.sums + a.pid[p] * 128 + 64 + n, lsq);
    }
  }
}

// ---------------------------------------------------------------------------
// Final W-conv: (128 x 256) x (256 x 64) per block; fused h' = o * tanh(temp).
// K processed in two 128-halves (co then mo), A+B halves staged per pass.
// ---------------------------------------------------------------------------
struct GemmFinalArgs {
  const unsigned short* co; const unsigned short* mo;
  const float* w; const float* bias; const float* o;
  float* hout;
};

__global__ __launch_bounds__(256) void stlstm_gemm_final(GemmFinalArgs a) {
  __shared__ unsigned short Bs[64 * LSTR];
  __shared__ unsigned short As[128 * LSTR];
  const int tid = threadIdx.x;
  const int n   = blockIdx.y;
  const int v0  = blockIdx.x * 64;

  const int wv = tid >> 5, lane = tid & 31;
  const int wm = (wv >> 1) * 32, wn = (wv & 1) * 32;
  const int l16 = lane & 15, lhi = lane >> 4;
  const v8f zero8 = {0.f, 0.f, 0.f, 0.f, 0.f, 0.f, 0.f, 0.f};

  v8f acc[2][2];
  #pragma unroll
  for (int tm = 0; tm < 2; ++tm)
    #pragma unroll
    for (int tn = 0; tn < 2; ++tn) acc[tm][tn] = zero8;

  for (int kh = 0; kh < 2; ++kh) {
    if (kh) __syncthreads();                  // previous half's reads done

    // Stage B half: k in [128*kh, 128*kh+128) from co (kh=0) or mo (kh=1).
    const unsigned short* ip =
        (kh ? a.mo : a.co) + (size_t)n * CV + v0;
    #pragma unroll
    for (int it = 0; it < 16; ++it) {
      int i = tid + it * 256;
      int v = i & 63, kp = i >> 6;
      unsigned short s0 = ip[(2 * kp) * V_LEN + v];
      unsigned short s1 = ip[(2 * kp + 1) * V_LEN + v];
      *(unsigned*)&Bs[v * LSTR + 2 * kp] = (unsigned)s0 | ((unsigned)s1 << 16);
    }
    // Stage A half: w_W columns [128*kh, 128*kh+128), rows 0..127.
    const float2* W2 = (const float2*)a.w;    // row = 128 float2 (256 floats)
    #pragma unroll
    for (int it = 0; it < 32; ++it) {
      int i = tid + it * 256;
      int m = i >> 6, kp = i & 63;
      float2 f = W2[m * 128 + kh * 64 + kp];
      *(unsigned*)&As[m * LSTR + 2 * kp] = pack2bf(f.x, f.y);
    }
    __syncthreads();

    #pragma unroll
    for (int ks = 0; ks < 4; ++ks) {
      FragU af[2], bf[2];
      #pragma unroll
      for (int t = 0; t < 2; ++t) {
        const unsigned short* ap = &As[(wm + t * 16 + l16) * LSTR + ks * 32 + lhi * 8];
        af[t].q[0] = *(const uint4*)(ap);
        af[t].q[1] = *(const uint4*)(ap + 16);
        const unsigned short* bp = &Bs[(wn + t * 16 + l16) * LSTR + ks * 32 + lhi * 8];
        bf[t].q[0] = *(const uint4*)(bp);
        bf[t].q[1] = *(const uint4*)(bp + 16);
      }
      #pragma unroll
      for (int tm = 0; tm < 2; ++tm)
        #pragma unroll
        for (int tn = 0; tn < 2; ++tn)
          acc[tm][tn] = __builtin_amdgcn_wmma_f32_16x16x32_bf16(
              false, af[tm].v, false, bf[tn].v, (short)0, acc[tm][tn], false, false);
    }
  }

  #pragma unroll
  for (int tm = 0; tm < 2; ++tm) {
    #pragma unroll
    for (int tn = 0; tn < 2; ++tn) {
      int col = v0 + wn + tn * 16 + l16;
      #pragma unroll
      for (int r = 0; r < 8; ++r) {
        int m = wm + tm * 16 + r + lhi * 8;
        size_t idx = (size_t)n * CV + m * V_LEN + col;
        float val = acc[tm][tn][r] + a.bias[m];
        a.hout[idx] = a.o[idx] * tanhf(val);
      }
    }
  }
}

// ---------------------------------------------------------------------------
// LN stat finalize: mu / rstd per (proj, sample)
// ---------------------------------------------------------------------------
__global__ void stlstm_ln_finalize(const float* sums, float* murs, int pstart) {
  int p = pstart + blockIdx.x;
  int n = threadIdx.x;
  float s = sums[p * 128 + n];
  float q = sums[p * 128 + 64 + n];
  float mu = s * (1.0f / 65536.0f);
  float var = q * (1.0f / 65536.0f) - mu * mu;
  murs[p * 128 + n] = mu;
  murs[p * 128 + 64 + n] = rsqrtf(var + 1e-5f);
}

__global__ void stlstm_zero(float* p, int n) {
  int i = blockIdx.x * 256 + threadIdx.x;
  if (i < n) p[i] = 0.f;
}

// ---------------------------------------------------------------------------
// Gate kernel: LN-apply 13 projections + temporal/spatial cell updates.
// Z order: xg,xi,xf,xg_,xi_,xf_,hg,hi,hf,mg,mi,mf,mm
// ---------------------------------------------------------------------------
struct GatesArgs {
  const unsigned short* Z[13];
  const float* g[13]; const float* be[13];
  const float* murs; int pid[13];
  const float* cin;
  float* cout; float* mout;
  unsigned short* cbf; unsigned short* mbf;
};

__global__ __launch_bounds__(256) void stlstm_gates(GatesArgs a) {
  unsigned e = blockIdx.x * 256 + threadIdx.x;
  int n  = (int)(e >> 16);
  int cv = (int)(e & 65535);
  float ln[13];
  #pragma unroll
  for (int j = 0; j < 13; ++j) {
    float mu = a.murs[a.pid[j] * 128 + n];
    float rs = a.murs[a.pid[j] * 128 + 64 + n];
    ln[j] = (bf2f(a.Z[j][e]) - mu) * rs * a.g[j][cv] + a.be[j][cv];
  }
  float g  = tanhf(ln[0] + ln[6]);
  float i  = sigm (ln[1] + ln[7]);
  float f  = sigm (ln[2] + ln[8] + 1.0f);
  float c_ = f * a.cin[e] + i * g;
  float g_ = tanhf(ln[3] + ln[9]);
  float i_ = sigm (ln[4] + ln[10]);
  float f_ = sigm (ln[5] + ln[11] + 1.0f);
  float m_ = f_ * ln[12] + i_ * g_;
  a.cout[e] = c_;  a.mout[e] = m_;
  a.cbf[e] = f2bf(c_);  a.mbf[e] = f2bf(m_);
}

// ---------------------------------------------------------------------------
// Output-gate kernel: Z order xo, ho, c, mo
// ---------------------------------------------------------------------------
struct OutGateArgs {
  const unsigned short* Z[4];
  const float* g[4]; const float* be[4];
  const float* murs; int pid[4];
  float* obuf; unsigned short* cobf; unsigned short* mobf;
};

__global__ __launch_bounds__(256) void stlstm_outgate(OutGateArgs a) {
  unsigned e = blockIdx.x * 256 + threadIdx.x;
  int n  = (int)(e >> 16);
  int cv = (int)(e & 65535);
  float ln[4];
  #pragma unroll
  for (int j = 0; j < 4; ++j) {
    float mu = a.murs[a.pid[j] * 128 + n];
    float rs = a.murs[a.pid[j] * 128 + 64 + n];
    ln[j] = (bf2f(a.Z[j][e]) - mu) * rs * a.g[j][cv] + a.be[j][cv];
  }
  float co = ln[2], mo = ln[3];
  a.obuf[e] = sigm(ln[0] + ln[1] + co + mo);
  a.cobf[e] = f2bf(co);
  a.mobf[e] = f2bf(mo);
}

// ---------------------------------------------------------------------------
// Host side
// ---------------------------------------------------------------------------
extern "C" void kernel_launch(void* const* d_in, const int* in_sizes, int n_in,
                              void* d_out, int out_size, void* d_ws, size_t ws_size,
                              hipStream_t stream) {
  (void)in_sizes; (void)n_in; (void)out_size; (void)ws_size;
  const float* x = (const float*)d_in[0];
  const float* h = (const float*)d_in[1];
  const float* c = (const float*)d_in[2];
  const float* m = (const float*)d_in[3];
  auto wptr  = [&](int j) { return (const float*)d_in[4 + 2 * j]; };
  auto bptr  = [&](int j) { return (const float*)d_in[5 + 2 * j]; };
  auto gptr  = [&](int k) { return (const float*)d_in[40 + 2 * k]; };
  auto beptr = [&](int k) { return (const float*)d_in[41 + 2 * k]; };

  // Z-slot order: xg,xi,xf,xg_,xi_,xf_,xo,hg,hi,hf,ho,mg,mi,mf,mm,c,mo
  // MAP -> index into _CONVS / _LNS ordering of the reference.
  static const int MAP[17] = {0,1,2,3,4,5,6,7,8,9,10,12,13,14,15,11,16};

  char* ws = (char*)d_ws;
  const size_t ZB = (size_t)NCV_SZ * 2;     // bf16 slab bytes
  unsigned short* Zs[17];
  for (int s = 0; s < 17; ++s) Zs[s] = (unsigned short*)(ws + (size_t)s * ZB);
  unsigned short* CBF  = (unsigned short*)(ws + 17 * ZB);
  unsigned short* MBF  = (unsigned short*)(ws + 18 * ZB);
  unsigned short* COBF = (unsigned short*)(ws + 19 * ZB);
  unsigned short* MOBF = (unsigned short*)(ws + 20 * ZB);
  float* OBUF = (float*)(ws + 21 * ZB);     // NCV floats (2 slots)
  float* SUMS = (float*)(ws + 23 * ZB);     // 17*128 floats
  float* MURS = SUMS + 17 * 128;

  float* hout = (float*)d_out;
  float* cout = hout + (size_t)NCV_SZ;
  float* mout = hout + 2 * (size_t)NCV_SZ;

  stlstm_zero<<<dim3((17 * 128 + 255) / 256), dim3(256), 0, stream>>>(SUMS, 17 * 128);

  dim3 gg(8, 64), gb(256);
  {   // x -> 7 projections
    GemmLnArgs A{}; A.inp = x; A.in_bf16 = 0; A.nproj = 7; A.sums = SUMS;
    for (int p = 0; p < 7; ++p) { int s = p;
      A.w[p] = wptr(MAP[s]); A.bias[p] = bptr(MAP[s]); A.z[p] = Zs[s]; A.pid[p] = s; }
    stlstm_gemm_ln<<<gg, gb, 0, stream>>>(A);
  }
  {   // h -> 4 projections
    GemmLnArgs A{}; A.inp = h; A.in_bf16 = 0; A.nproj = 4; A.sums = SUMS;
    for (int p = 0; p < 4; ++p) { int s = 7 + p;
      A.w[p] = wptr(MAP[s]); A.bias[p] = bptr(MAP[s]); A.z[p] = Zs[s]; A.pid[p] = s; }
    stlstm_gemm_ln<<<gg, gb, 0, stream>>>(A);
  }
  {   // m -> 4 projections
    GemmLnArgs A{}; A.inp = m; A.in_bf16 = 0; A.nproj = 4; A.sums = SUMS;
    for (int p = 0; p < 4; ++p) { int s = 11 + p;
      A.w[p] = wptr(MAP[s]); A.bias[p] = bptr(MAP[s]); A.z[p] = Zs[s]; A.pid[p] = s; }
    stlstm_gemm_ln<<<gg, gb, 0, stream>>>(A);
  }
  stlstm_ln_finalize<<<dim3(15), dim3(64), 0, stream>>>(SUMS, MURS, 0);

  {   // gates -> c_, m_
    GatesArgs G{};
    static const int ZS[13] = {0,1,2,3,4,5,7,8,9,11,12,13,14};
    for (int j = 0; j < 13; ++j) { int s = ZS[j];
      G.Z[j] = Zs[s]; G.g[j] = gptr(MAP[s]); G.be[j] = beptr(MAP[s]); G.pid[j] = s; }
    G.murs = MURS; G.cin = c; G.cout = cout; G.mout = mout; G.cbf = CBF; G.mbf = MBF;
    stlstm_gates<<<dim3(NCV_SZ / 256), gb, 0, stream>>>(G);
  }

  {   // c_ -> 'c' projection
    GemmLnArgs A{}; A.inp = CBF; A.in_bf16 = 1; A.nproj = 1; A.sums = SUMS;
    int s = 15; A.w[0] = wptr(MAP[s]); A.bias[0] = bptr(MAP[s]); A.z[0] = Zs[s]; A.pid[0] = s;
    stlstm_gemm_ln<<<gg, gb, 0, stream>>>(A);
  }
  {   // m_ -> 'mo' projection
    GemmLnArgs A{}; A.inp = MBF; A.in_bf16 = 1; A.nproj = 1; A.sums = SUMS;
    int s = 16; A.w[0] = wptr(MAP[s]); A.bias[0] = bptr(MAP[s]); A.z[0] = Zs[s]; A.pid[0] = s;
    stlstm_gemm_ln<<<gg, gb, 0, stream>>>(A);
  }
  stlstm_ln_finalize<<<dim3(2), dim3(64), 0, stream>>>(SUMS, MURS, 15);

  {   // output gate + co/mo bf16 for W conv
    OutGateArgs O{};
    static const int ZS4[4] = {6, 10, 15, 16};
    for (int j = 0; j < 4; ++j) { int s = ZS4[j];
      O.Z[j] = Zs[s]; O.g[j] = gptr(MAP[s]); O.be[j] = beptr(MAP[s]); O.pid[j] = s; }
    O.murs = MURS; O.obuf = OBUF; O.cobf = COBF; O.mobf = MOBF;
    stlstm_outgate<<<dim3(NCV_SZ / 256), gb, 0, stream>>>(O);
  }

  {   // W conv (K=256) + fused h' = o * tanh(temp)
    GemmFinalArgs F{};
    F.co = COBF; F.mo = MOBF; F.w = wptr(17); F.bias = bptr(17); F.o = OBUF; F.hout = hout;
    stlstm_gemm_final<<<gg, gb, 0, stream>>>(F);
  }
}
